// ScaledDotProductWithEdgeAttention_49220325212255
// MI455X (gfx1250) — matrix operand order repaired and checked by reading, not compiled, not run on hardware
//
#include <hip/hip_runtime.h>
#include <hip/hip_bf16.h>

typedef _Float16 v4h  __attribute__((ext_vector_type(4)));
typedef _Float16 v8h  __attribute__((ext_vector_type(8)));
typedef _Float16 v16h __attribute__((ext_vector_type(16)));
typedef float    v8f  __attribute__((ext_vector_type(8)));

#define LQ 128
#define LK 128
#define DD 64
#define QK_STRIDE 72    // 64 + 8 halves pad -> 144B rows (16B aligned, spreads banks)
#define VT_STRIDE 136   // 128 + 8 halves pad -> 272B rows
#define P_STRIDE  136

__global__ __launch_bounds__(256)
void attn_fused_kernel(const float* __restrict__ q, const float* __restrict__ k,
                       const float* __restrict__ v, const int* __restrict__ mask,
                       float* __restrict__ out)
{
    __shared__ _Float16 Qs[LQ * QK_STRIDE];       // Q/8 as f16, row-major
    __shared__ _Float16 Ks[LK * QK_STRIDE];       // K as f16, row-major
    __shared__ _Float16 Vt[DD * VT_STRIDE];       // V transposed: [d][k]
    __shared__ _Float16 Ps[8 * 16 * P_STRIDE];    // per-wave softmax rows

    const int tid  = threadIdx.x;
    const int bhn  = blockIdx.x;                  // b*H*N + h*N + n
    const size_t tile = (size_t)bhn * (LQ * DD);
    const int bIdx = bhn >> 10;                   // / (H*N) = /1024
    const int nIdx = bhn & 127;
    const int* __restrict__ mbase = mask + ((size_t)(bIdx * 128 + nIdx)) * (LQ * LK);

    // ---- stage Q (pre-scaled), K row-major; V transposed; all f16, coalesced 16B global loads ----
    #pragma unroll
    for (int it = 0; it < 8; ++it) {
        int g   = tid + it * 256;                 // group of 4 consecutive floats
        int e   = g * 4;
        int row = e >> 6;
        int col = e & 63;
        float4 qv = *(const float4*)(q + tile + e);
        float4 kv = *(const float4*)(k + tile + e);
        float4 vv = *(const float4*)(v + tile + e);
        v4h qh = { (_Float16)(qv.x * 0.125f), (_Float16)(qv.y * 0.125f),
                   (_Float16)(qv.z * 0.125f), (_Float16)(qv.w * 0.125f) };
        v4h kh = { (_Float16)kv.x, (_Float16)kv.y, (_Float16)kv.z, (_Float16)kv.w };
        *(v4h*)&Qs[row * QK_STRIDE + col] = qh;
        *(v4h*)&Ks[row * QK_STRIDE + col] = kh;
        Vt[(col + 0) * VT_STRIDE + row] = (_Float16)vv.x;
        Vt[(col + 1) * VT_STRIDE + row] = (_Float16)vv.y;
        Vt[(col + 2) * VT_STRIDE + row] = (_Float16)vv.z;
        Vt[(col + 3) * VT_STRIDE + row] = (_Float16)vv.w;
    }
    __syncthreads();

    const int lane  = tid & 31;
    const int wave  = tid >> 5;
    const int mcol  = lane & 15;     // N / M-row index within fragments
    const int hf    = lane >> 4;     // half-wave group
    const int qrow0 = wave * 16;

    // ---- A fragments of Q (16x32 f16 layout: contiguous run at +16*hf) ----
    v16h aq[2];
    #pragma unroll
    for (int dc = 0; dc < 2; ++dc) {
        const _Float16* p = &Qs[(qrow0 + mcol) * QK_STRIDE + dc * 32 + 16 * hf];
        ((v8h*)&aq[dc])[0] = *(const v8h*)p;
        ((v8h*)&aq[dc])[1] = *(const v8h*)(p + 8);
    }

    // ---- S = (Q/8) @ K^T : 8 column blocks x 2 d-chunks of WMMA ----
    v8f acc[8];
    #pragma unroll
    for (int blk = 0; blk < 8; ++blk) {
        v8f c = {};
        #pragma unroll
        for (int dc = 0; dc < 2; ++dc) {
            const _Float16* p = &Ks[(blk * 16 + mcol) * QK_STRIDE + dc * 32 + 16 * hf];
            v16h b;
            ((v8h*)&b)[0] = *(const v8h*)p;
            ((v8h*)&b)[1] = *(const v8h*)(p + 8);
            c = __builtin_amdgcn_wmma_f32_16x16x32_f16(false, aq[dc], false, b,
                                                       (short)0, c, false, false);
        }
        acc[blk] = c;
    }

    // ---- mask + row-wise softmax (rows live in 16-lane half-groups) ----
    float rmax[8], rsum[8];
    #pragma unroll
    for (int j = 0; j < 8; ++j) rmax[j] = -3.0e38f;
    #pragma unroll
    for (int blk = 0; blk < 8; ++blk) {
        #pragma unroll
        for (int j = 0; j < 8; ++j) {
            int mrow = qrow0 + j + 8 * hf;
            int mk = mbase[mrow * LK + blk * 16 + mcol];
            float s = (mk == 0) ? -32768.0f : acc[blk][j];
            acc[blk][j] = s;
            rmax[j] = fmaxf(rmax[j], s);
        }
    }
    #pragma unroll
    for (int j = 0; j < 8; ++j) {
        #pragma unroll
        for (int x = 1; x < 16; x <<= 1)
            rmax[j] = fmaxf(rmax[j], __shfl_xor(rmax[j], x, 32));
        rsum[j] = 0.0f;
    }
    #pragma unroll
    for (int blk = 0; blk < 8; ++blk) {
        #pragma unroll
        for (int j = 0; j < 8; ++j) {
            float e = __expf(acc[blk][j] - rmax[j]);
            acc[blk][j] = e;
            rsum[j] += e;
        }
    }
    #pragma unroll
    for (int j = 0; j < 8; ++j) {
        #pragma unroll
        for (int x = 1; x < 16; x <<= 1)
            rsum[j] += __shfl_xor(rsum[j], x, 32);
        rsum[j] = 1.0f / rsum[j];
    }

    // ---- write normalized P (f16) to this wave's LDS strip (same-wave LDS is in-order) ----
    _Float16* pw = &Ps[wave * 16 * P_STRIDE];
    #pragma unroll
    for (int blk = 0; blk < 8; ++blk) {
        #pragma unroll
        for (int j = 0; j < 8; ++j)
            pw[(j + 8 * hf) * P_STRIDE + blk * 16 + mcol] =
                (_Float16)(acc[blk][j] * rsum[j]);
    }

    // ---- O = P @ V : 4 k-chunks x 4 d-blocks of WMMA (Vt gives contiguous-K B frags) ----
    v8f oacc[4];
    #pragma unroll
    for (int db = 0; db < 4; ++db) oacc[db] = (v8f){};
    #pragma unroll
    for (int kc = 0; kc < 4; ++kc) {
        const _Float16* pa = &pw[mcol * P_STRIDE + kc * 32 + 16 * hf];
        v16h ap;
        ((v8h*)&ap)[0] = *(const v8h*)pa;
        ((v8h*)&ap)[1] = *(const v8h*)(pa + 8);
        #pragma unroll
        for (int db = 0; db < 4; ++db) {
            const _Float16* pb = &Vt[(db * 16 + mcol) * VT_STRIDE + kc * 32 + 16 * hf];
            v16h b;
            ((v8h*)&b)[0] = *(const v8h*)pb;
            ((v8h*)&b)[1] = *(const v8h*)(pb + 8);
            oacc[db] = __builtin_amdgcn_wmma_f32_16x16x32_f16(false, ap, false, b,
                                                              (short)0, oacc[db], false, false);
        }
    }

    // ---- coalesced f32 stores ----
    float* ob = out + tile;
    #pragma unroll
    for (int db = 0; db < 4; ++db) {
        #pragma unroll
        for (int j = 0; j < 8; ++j)
            ob[(qrow0 + j + 8 * hf) * DD + db * 16 + mcol] = oacc[db][j];
    }
}

extern "C" void kernel_launch(void* const* d_in, const int* in_sizes, int n_in,
                              void* d_out, int out_size, void* d_ws, size_t ws_size,
                              hipStream_t stream) {
    (void)in_sizes; (void)n_in; (void)out_size; (void)d_ws; (void)ws_size;
    const float* q    = (const float*)d_in[0];
    const float* k    = (const float*)d_in[1];
    const float* v    = (const float*)d_in[2];
    const int*   mask = (const int*)d_in[3];
    float* out = (float*)d_out;
    // 4*8*128 = 4096 independent attention problems, one workgroup each
    attn_fused_kernel<<<dim3(4096), dim3(256), 0, stream>>>(q, k, v, mask, out);
}
